// CamLiRAFT_L_Core_83786222010979
// MI455X (gfx1250) — matrix operand order repaired and checked by reading, not compiled, not run on hardware
//
#include <hip/hip_runtime.h>
#include <math.h>

// ---------------------------------------------------------------------------
// CDNA5 (gfx1250) CamLiRAFT-L core. Dense channel mixing via
// v_wmma_f32_16x16x32_f16 on pre-packed zero-padded f16 operands. Each wave
// computes a 32x32 output macro-tile: 2 A-frags + 2 B-frags -> 4 wmma.
// ---------------------------------------------------------------------------

typedef __attribute__((ext_vector_type(16))) _Float16 v16h;
typedef __attribute__((ext_vector_type(8)))  _Float16 v8h;
typedef __attribute__((ext_vector_type(8)))  float    v8f;

#define NB 2  // batch

// ---------------- activation -------------------------------------------------
template<int ACT>
__device__ __forceinline__ float apply_act(float v) {
  if (ACT == 1) return fmaxf(v, 0.f);                 // relu
  if (ACT == 2) return v < 0.f ? 0.1f * v : v;        // leaky relu 0.1
  if (ACT == 3) return tanhf(v);                      // tanh
  if (ACT == 4) return 1.f / (1.f + expf(-v));        // sigmoid
  return v;
}

// ---------------- operand packing -------------------------------------------
// activations: (B,C,N) f32 -> (B,N,Cp) f16, zero padded to Cp (mult of 32)
__global__ void pack_x_kernel(const float* __restrict__ x, int C, int N, int Cp,
                              _Float16* __restrict__ xh) {
  long total = (long)NB * N * Cp;
  long gid = (long)blockIdx.x * blockDim.x + threadIdx.x;
  if (gid >= total) return;
  int c = (int)(gid % Cp);
  int n = (int)((gid / Cp) % N);
  int b = (int)(gid / ((long)N * Cp));
  float v = (c < C) ? x[((size_t)b * C + c) * N + n] : 0.f;
  xh[gid] = (_Float16)v;
}
// weights: (O,C) f32 -> (Op,Cp) f16, zero padded (Op mult of 32)
__global__ void pack_w_kernel(const float* __restrict__ w, int C, int O,
                              int Cp, int Op, _Float16* __restrict__ wh) {
  long total = (long)Op * Cp;
  long gid = (long)blockIdx.x * blockDim.x + threadIdx.x;
  if (gid >= total) return;
  int c = (int)(gid % Cp);
  int o = (int)(gid / Cp);
  wh[gid] = (_Float16)((o < O && c < C) ? w[(size_t)o * C + c] : 0.f);
}

__device__ __forceinline__ v16h load_a_frag(const _Float16* __restrict__ row,
                                            int k0, int half) {
  // A 16x32 f16 fragment: K = 16*half + h  -> one contiguous 16-half run
  return *(const v16h*)(row + k0 + half * 16);
}

// ---------------- WMMA 1x1 conv on packed operands (32x32 tile / wave) -------
// out[b,o,n] = act(sum_c W[o,c]*X[b,c,n] + bias[o]); wh (Op,Cp), xh (B,N,Cp)
template<int ACT>
__global__ void conv1_wmma_kernel(const _Float16* __restrict__ wh,
                                  const float* __restrict__ bias,
                                  const _Float16* __restrict__ xh,
                                  float* __restrict__ out,
                                  int Cp, int N, int O) {
  const int lane  = threadIdx.x;
  const int ntile = blockIdx.x, otile = blockIdx.y, b = blockIdx.z;
  const int l15  = lane & 15;
  const int half = lane >> 4;
  const int ncol0 = ntile * 32 + l15, ncol1 = ncol0 + 16;
  const int arow0 = otile * 32 + l15, arow1 = arow0 + 16;
  const _Float16* wrow0 = wh + (size_t)arow0 * Cp;
  const _Float16* wrow1 = wh + (size_t)arow1 * Cp;
  const _Float16* xrow0 = xh + ((size_t)b * N + ncol0) * Cp;
  const _Float16* xrow1 = xh + ((size_t)b * N + ncol1) * Cp;
  v8f acc00 = {}, acc01 = {}, acc10 = {}, acc11 = {};
  for (int k0 = 0; k0 < Cp; k0 += 32) {
    v16h a0 = load_a_frag(wrow0, k0, half);
    v16h a1 = load_a_frag(wrow1, k0, half);
    v16h b0 = *(const v16h*)(xrow0 + k0 + half * 16);  // B: K=16*half+h
    v16h b1 = *(const v16h*)(xrow1 + k0 + half * 16);
    acc00 = __builtin_amdgcn_wmma_f32_16x16x32_f16(false, a0, false, b0, (short)0, acc00, false, false);
    acc01 = __builtin_amdgcn_wmma_f32_16x16x32_f16(false, a0, false, b1, (short)0, acc01, false, false);
    acc10 = __builtin_amdgcn_wmma_f32_16x16x32_f16(false, a1, false, b0, (short)0, acc10, false, false);
    acc11 = __builtin_amdgcn_wmma_f32_16x16x32_f16(false, a1, false, b1, (short)0, acc11, false, false);
  }
#pragma unroll
  for (int r = 0; r < 8; ++r) {      // C/D: M = r + 8*half, N = l15
    int o0 = otile * 32 + r + 8 * half;
    int o1 = o0 + 16;
    if (o0 < O) {
      float bb = bias[o0];
      out[((size_t)b * O + o0) * N + ncol0] = apply_act<ACT>(acc00[r] + bb);
      out[((size_t)b * O + o0) * N + ncol1] = apply_act<ACT>(acc01[r] + bb);
    }
    if (o1 < O) {
      float bb = bias[o1];
      out[((size_t)b * O + o1) * N + ncol0] = apply_act<ACT>(acc10[r] + bb);
      out[((size_t)b * O + o1) * N + ncol1] = apply_act<ACT>(acc11[r] + bb);
    }
  }
}

// ---------------- WMMA correlation volume on packed feats (32x32 tile) -------
// cv[b,n,m] = scale * sum_c f1[b,c,n]*f2[b,c,m]; inputs packed (B,Npts,Cp)
__global__ void corrvol_wmma_kernel(const _Float16* __restrict__ Xh,
                                    const _Float16* __restrict__ Yh,
                                    int Nn, int Nm, int Cp, float scale,
                                    float* __restrict__ out) {
  const int lane  = threadIdx.x;
  const int mtile = blockIdx.x, ntile = blockIdx.y, b = blockIdx.z;
  const int l15  = lane & 15;
  const int half = lane >> 4;
  const int nrow0 = ntile * 32 + l15, nrow1 = nrow0 + 16;
  const int mcol0 = mtile * 32 + l15, mcol1 = mcol0 + 16;
  const _Float16* arow0 = Xh + ((size_t)b * Nn + nrow0) * Cp;
  const _Float16* arow1 = Xh + ((size_t)b * Nn + nrow1) * Cp;
  const _Float16* brow0 = Yh + ((size_t)b * Nm + mcol0) * Cp;
  const _Float16* brow1 = Yh + ((size_t)b * Nm + mcol1) * Cp;
  v8f acc00 = {}, acc01 = {}, acc10 = {}, acc11 = {};
  for (int k0 = 0; k0 < Cp; k0 += 32) {
    v16h a0 = load_a_frag(arow0, k0, half);
    v16h a1 = load_a_frag(arow1, k0, half);
    v16h b0 = *(const v16h*)(brow0 + k0 + half * 16);
    v16h b1 = *(const v16h*)(brow1 + k0 + half * 16);
    acc00 = __builtin_amdgcn_wmma_f32_16x16x32_f16(false, a0, false, b0, (short)0, acc00, false, false);
    acc01 = __builtin_amdgcn_wmma_f32_16x16x32_f16(false, a0, false, b1, (short)0, acc01, false, false);
    acc10 = __builtin_amdgcn_wmma_f32_16x16x32_f16(false, a1, false, b0, (short)0, acc10, false, false);
    acc11 = __builtin_amdgcn_wmma_f32_16x16x32_f16(false, a1, false, b1, (short)0, acc11, false, false);
  }
#pragma unroll
  for (int r = 0; r < 8; ++r) {
    int n0 = ntile * 32 + r + 8 * half;
    int n1 = n0 + 16;
    out[((size_t)b * Nn + n0) * Nm + mcol0] = acc00[r] * scale;
    out[((size_t)b * Nn + n0) * Nm + mcol1] = acc01[r] * scale;
    out[((size_t)b * Nn + n1) * Nm + mcol0] = acc10[r] * scale;
    out[((size_t)b * Nn + n1) * Nm + mcol1] = acc11[r] * scale;
  }
}

// ---------------- KNN top-K (smallest squared distance) ----------------------
template<int K>
__global__ void knn_kernel(const float* __restrict__ src, int Ns,
                           const float* __restrict__ qry, int Nq,
                           int* __restrict__ out) {
  int gid = blockIdx.x * blockDim.x + threadIdx.x;
  if (gid >= NB * Nq) return;
  int b = gid / Nq, m = gid % Nq;
  float qx = qry[(b * 3 + 0) * Nq + m];
  float qy = qry[(b * 3 + 1) * Nq + m];
  float qz = qry[(b * 3 + 2) * Nq + m];
  float d[K]; int id[K];
#pragma unroll
  for (int t = 0; t < K; ++t) { d[t] = 3.0e38f; id[t] = 0; }
  for (int j = 0; j < Ns; ++j) {
    float dx = src[(b * 3 + 0) * Ns + j] - qx;
    float dy = src[(b * 3 + 1) * Ns + j] - qy;
    float dz = src[(b * 3 + 2) * Ns + j] - qz;
    float dist = dx * dx + dy * dy + dz * dz;
    if (dist < d[K - 1]) {
      float cd = dist; int ci = j;
#pragma unroll
      for (int t = 0; t < K; ++t) {
        if (cd < d[t]) { float td = d[t]; int ti = id[t];
                         d[t] = cd; id[t] = ci; cd = td; ci = ti; }
      }
    }
  }
#pragma unroll
  for (int t = 0; t < K; ++t) out[(b * Nq + m) * K + t] = id[t];
}

// ---------------- KNN3 inverse-distance interpolation (+optional base-minus) -
__global__ void knn3_interp_kernel(const float* __restrict__ src, int Ns,
                                   const float* __restrict__ tgt, int Nt,
                                   const float* __restrict__ feat, int C,
                                   const float* __restrict__ base,
                                   float* __restrict__ out) {
  int gid = blockIdx.x * blockDim.x + threadIdx.x;
  if (gid >= NB * Nt) return;
  int b = gid / Nt, t = gid % Nt;
  float qx = tgt[(b * 3 + 0) * Nt + t];
  float qy = tgt[(b * 3 + 1) * Nt + t];
  float qz = tgt[(b * 3 + 2) * Nt + t];
  float d0 = 3e38f, d1 = 3e38f, d2 = 3e38f; int i0 = 0, i1 = 0, i2 = 0;
  for (int j = 0; j < Ns; ++j) {
    float dx = src[(b * 3 + 0) * Ns + j] - qx;
    float dy = src[(b * 3 + 1) * Ns + j] - qy;
    float dz = src[(b * 3 + 2) * Ns + j] - qz;
    float dist = dx * dx + dy * dy + dz * dz;
    if (dist < d2) {
      if (dist < d0)      { d2 = d1; i2 = i1; d1 = d0; i1 = i0; d0 = dist; i0 = j; }
      else if (dist < d1) { d2 = d1; i2 = i1; d1 = dist; i1 = j; }
      else                { d2 = dist; i2 = j; }
    }
  }
  float w0 = 1.f / (sqrtf(fmaxf(d0, 0.f)) + 1e-8f);
  float w1 = 1.f / (sqrtf(fmaxf(d1, 0.f)) + 1e-8f);
  float w2 = 1.f / (sqrtf(fmaxf(d2, 0.f)) + 1e-8f);
  float ws = w0 + w1 + w2; w0 /= ws; w1 /= ws; w2 /= ws;
  for (int c = 0; c < C; ++c) {
    float v = w0 * feat[(b * C + c) * Ns + i0] +
              w1 * feat[(b * C + c) * Ns + i1] +
              w2 * feat[(b * C + c) * Ns + i2];
    if (base) v = base[(b * 3 + c) * Nt + t] - v;
    out[(b * C + c) * Nt + t] = v;
  }
}

// ---------------- PointConv: wk[b,m,k,w] = relu(Wn(rel)+b) -------------------
__global__ void pconv_wk_kernel(const float* __restrict__ src, int Ns,
                                const float* __restrict__ qry, int Nq,
                                const int* __restrict__ idx,
                                const float* __restrict__ wnw,
                                const float* __restrict__ wnb,
                                float* __restrict__ wk) {
  int gid = blockIdx.x * blockDim.x + threadIdx.x;
  if (gid >= NB * Nq * 16) return;
  int b = gid / (Nq * 16); int rem = gid % (Nq * 16);
  int m = rem >> 4, k = rem & 15;
  int j = idx[(b * Nq + m) * 16 + k];
  float r0 = src[(b * 3 + 0) * Ns + j] - qry[(b * 3 + 0) * Nq + m];
  float r1 = src[(b * 3 + 1) * Ns + j] - qry[(b * 3 + 1) * Nq + m];
  float r2 = src[(b * 3 + 2) * Ns + j] - qry[(b * 3 + 2) * Nq + m];
#pragma unroll
  for (int w = 0; w < 16; ++w) {
    float v = wnw[w * 3 + 0] * r0 + wnw[w * 3 + 1] * r1 + wnw[w * 3 + 2] * r2 + wnb[w];
    wk[((size_t)(b * Nq + m) * 16 + k) * 16 + w] = fmaxf(v, 0.f);
  }
}

// agg[b, c*16+w, m] = sum_k feat[b,c,idx[m,k]] * wk[b,m,k,w]
__global__ void pconv_agg_kernel(const float* __restrict__ feat, int C, int Ns,
                                 const int* __restrict__ idx, int Nq,
                                 const float* __restrict__ wk,
                                 float* __restrict__ agg) {
  int C16 = C * 16;
  long total = (long)NB * C16 * Nq;
  long gid = (long)blockIdx.x * blockDim.x + threadIdx.x;
  if (gid >= total) return;
  int b = (int)(gid / ((long)C16 * Nq));
  long rem = gid % ((long)C16 * Nq);
  int i = (int)(rem / Nq), m = (int)(rem % Nq);
  int c = i >> 4, w = i & 15;
  float acc = 0.f;
  for (int k = 0; k < 16; ++k) {
    int j = idx[(b * Nq + m) * 16 + k];
    acc += feat[(b * C + c) * Ns + j] * wk[((size_t)(b * Nq + m) * 16 + k) * 16 + w];
  }
  agg[((size_t)b * C16 + i) * Nq + m] = acc;
}

// ---------------- depthwise pconv (fixed knn of xyz1) ------------------------
__global__ void relpre_kernel(const float* __restrict__ xyz,
                              const int* __restrict__ knn32,
                              float* __restrict__ rel, int N) {
  int gid = blockIdx.x * blockDim.x + threadIdx.x;
  if (gid >= NB * N * 32) return;
  int b = gid / (N * 32); int r = gid % (N * 32);
  int m = r / 32, k = r % 32;
  int j = knn32[(b * N + m) * 32 + k];
#pragma unroll
  for (int d = 0; d < 3; ++d)
    rel[((size_t)(b * N + m) * 32 + k) * 3 + d] =
        xyz[(b * 3 + d) * N + j] - xyz[(b * 3 + d) * N + m];
}

__global__ void dw_s_kernel(const float* __restrict__ rel,
                            const int* __restrict__ knn32,
                            const float* __restrict__ feat,
                            const float* __restrict__ wnw,
                            const float* __restrict__ wnb,
                            int C, int kuse, int N, float* __restrict__ s) {
  long total = (long)NB * C * N;
  long gid = (long)blockIdx.x * blockDim.x + threadIdx.x;
  if (gid >= total) return;
  int b = (int)(gid / ((long)C * N));
  long r = gid % ((long)C * N);
  int c = (int)(r / N), m = (int)(r % N);
  float w0 = wnw[c * 3 + 0], w1 = wnw[c * 3 + 1], w2 = wnw[c * 3 + 2], wb = wnb[c];
  float acc = 0.f;
  for (int k = 0; k < kuse; ++k) {
    const float* rp = &rel[((size_t)(b * N + m) * 32 + k) * 3];
    float wv = fmaxf(w0 * rp[0] + w1 * rp[1] + w2 * rp[2] + wb, 0.f);
    int j = knn32[(b * N + m) * 32 + k];
    acc += wv * feat[(b * C + c) * N + j];
  }
  s[(b * C + c) * N + m] = acc / (float)kuse;
}

// ---------------- pyramid avg: out[b,n,m]=mean_3 prev[b,n,idx3[m,:]] ---------
__global__ void pyr_avg_kernel(const float* __restrict__ prev, int Lp,
                               const int* __restrict__ idx3, int Lc,
                               float* __restrict__ out) {
  long total = (long)NB * 2048 * Lc;
  long gid = (long)blockIdx.x * blockDim.x + threadIdx.x;
  if (gid >= total) return;
  int b = (int)(gid / ((long)2048 * Lc));
  long r = gid % ((long)2048 * Lc);
  int n = (int)(r / Lc), m = (int)(r % Lc);
  const int* ip = &idx3[(b * Lc + m) * 3];
  const float* row = &prev[((size_t)b * 2048 + n) * Lp];
  out[((size_t)b * 2048 + n) * Lc + m] = (row[ip[0]] + row[ip[1]] + row[ip[2]]) * (1.f / 3.f);
}

// ---------------- fused cost MLP: 4 -> 32 -> 32, sum over 16 neighbors -------
__global__ void cost_kernel(const float* __restrict__ xyz1,
                            const float* __restrict__ xyz2, int Ni,
                            const int* __restrict__ idx,
                            const float* __restrict__ cv,
                            const float* __restrict__ W1, const float* __restrict__ b1,
                            const float* __restrict__ W2, const float* __restrict__ b2,
                            int lvl, float* __restrict__ cost) {
  __shared__ float sW1[128], sb1[32], sW2[1024], sb2[32];
  for (int i = threadIdx.x; i < 128;  i += blockDim.x) sW1[i] = W1[i];
  for (int i = threadIdx.x; i < 32;   i += blockDim.x) { sb1[i] = b1[i]; sb2[i] = b2[i]; }
  for (int i = threadIdx.x; i < 1024; i += blockDim.x) sW2[i] = W2[i];
  __syncthreads();
  int gid = blockIdx.x * blockDim.x + threadIdx.x;
  if (gid >= NB * 2048) return;
  int b = gid / 2048, m = gid % 2048;
  float qx = xyz1[(b * 3 + 0) * 2048 + m];
  float qy = xyz1[(b * 3 + 1) * 2048 + m];
  float qz = xyz1[(b * 3 + 2) * 2048 + m];
  float acc[32];
#pragma unroll
  for (int o = 0; o < 32; ++o) acc[o] = 0.f;
  for (int k = 0; k < 16; ++k) {
    int j = idx[(b * 2048 + m) * 16 + k];
    float in0 = xyz2[(b * 3 + 0) * Ni + j] - qx;
    float in1 = xyz2[(b * 3 + 1) * Ni + j] - qy;
    float in2 = xyz2[(b * 3 + 2) * Ni + j] - qz;
    float in3 = cv[((size_t)b * 2048 + m) * Ni + j];
    float h1[32];
#pragma unroll
    for (int o = 0; o < 32; ++o) {
      float v = sW1[o * 4 + 0] * in0 + sW1[o * 4 + 1] * in1 +
                sW1[o * 4 + 2] * in2 + sW1[o * 4 + 3] * in3 + sb1[o];
      h1[o] = fmaxf(v, 0.f);
    }
#pragma unroll
    for (int o = 0; o < 32; ++o) {
      float v = sb2[o];
#pragma unroll
      for (int jj = 0; jj < 32; ++jj) v += sW2[o * 32 + jj] * h1[jj];
      acc[o] += fmaxf(v, 0.f);
    }
  }
#pragma unroll
  for (int o = 0; o < 32; ++o)
    cost[(b * 128 + lvl * 32 + o) * 2048 + m] = acc[o];
}

// ---------------- elementwise glue ------------------------------------------
__global__ void slice_xyz_kernel(const float* __restrict__ src, int nstride,
                                 int n, float* __restrict__ dst) {
  int gid = blockIdx.x * blockDim.x + threadIdx.x;
  if (gid >= NB * 3 * n) return;
  int bd = gid / n, i = gid % n;
  dst[bd * n + i] = src[bd * nstride + i];
}
__global__ void zero_kernel(float* p, int n) {
  int gid = blockIdx.x * blockDim.x + threadIdx.x;
  if (gid < n) p[gid] = 0.f;
}
__global__ void add_kernel(float* a, const float* b, int n) {
  int gid = blockIdx.x * blockDim.x + threadIdx.x;
  if (gid < n) a[gid] += b[gid];
}
__global__ void copy_ch_kernel(float* __restrict__ dst, int Cd, int doff,
                               const float* __restrict__ src, int Cs, int soff,
                               int Cc, int N) {
  long total = (long)NB * Cc * N;
  long gid = (long)blockIdx.x * blockDim.x + threadIdx.x;
  if (gid >= total) return;
  int b = (int)(gid / ((long)Cc * N));
  long r = gid % ((long)Cc * N);
  int c = (int)(r / N), n = (int)(r % N);
  dst[(b * Cd + doff + c) * N + n] = src[(b * Cs + soff + c) * N + n];
}
__global__ void split_featc_kernel(const float* __restrict__ fc,
                                   float* __restrict__ h, float* __restrict__ x, int N) {
  long total = (long)NB * 128 * N;
  long gid = (long)blockIdx.x * blockDim.x + threadIdx.x;
  if (gid >= total) return;
  int b = (int)(gid / (128L * N));
  long r = gid % (128L * N);
  int c = (int)(r / N), n = (int)(r % N);
  h[(b * 128 + c) * N + n] = tanhf(fc[(b * 256 + c) * N + n]);
  x[(b * 128 + c) * N + n] = fmaxf(fc[(b * 256 + 128 + c) * N + n], 0.f);
}
__global__ void rh_kernel(const float* __restrict__ rg, const float* __restrict__ h,
                          float* __restrict__ qin, int N) {
  long total = (long)NB * 128 * N;
  long gid = (long)blockIdx.x * blockDim.x + threadIdx.x;
  if (gid >= total) return;
  int b = (int)(gid / (128L * N));
  long r = gid % (128L * N);
  int c = (int)(r / N), n = (int)(r % N);
  qin[(b * 384 + c) * N + n] = rg[(b * 128 + c) * N + n] * h[(b * 128 + c) * N + n];
}
__global__ void gru_combine_kernel(float* __restrict__ h, const float* __restrict__ z,
                                   const float* __restrict__ q, int N) {
  long total = (long)NB * 128 * N;
  long gid = (long)blockIdx.x * blockDim.x + threadIdx.x;
  if (gid >= total) return;
  float zz = z[gid];
  h[gid] = (1.f - zz) * h[gid] + zz * q[gid];
}

// ---------------------------------------------------------------------------
// Host orchestration
// ---------------------------------------------------------------------------
struct Dn { const float* w; const float* b; };
struct Pc { Dn wn; Dn lin; };
struct Dw { Dn wn; Dn out; };
struct EncW { Dn l0[2]; Dn m0[2]; Dn m1[2]; Pc c0; Pc c1; };

static inline int nblk(long total, int bs) { return (int)((total + bs - 1) / bs); }

extern "C" void kernel_launch(void* const* d_in, const int* in_sizes, int n_in,
                              void* d_out, int out_size, void* d_ws, size_t ws_size,
                              hipStream_t stream) {
  if (n_in < 88) return;
  (void)in_sizes; (void)out_size; (void)ws_size;

  const float* pc1 = (const float*)d_in[0];
  const float* pc2 = (const float*)d_in[1];

  // ---- unpack params (setup_inputs insertion order) ----
  int qi = 2;
  auto NXT = [&]() { return (const float*)d_in[qi++]; };
  auto rdDn = [&]() { Dn d; d.w = NXT(); d.b = NXT(); return d; };
  auto rdEnc = [&]() {
    EncW e;
    e.l0[0] = rdDn(); e.l0[1] = rdDn();
    e.m0[0] = rdDn(); e.m0[1] = rdDn();
    e.m1[0] = rdDn(); e.m1[1] = rdDn();
    e.c0.wn = rdDn(); e.c0.lin = rdDn();
    e.c1.wn = rdDn(); e.c1.lin = rdDn();
    return e;
  };
  auto rdDw = [&]() { Dw d; d.wn = rdDn(); d.out = rdDn(); return d; };

  EncW fnet = rdEnc();
  EncW cnet = rdEnc();
  Dn aligner = rdDn();
  Dn cm0 = rdDn(), cm1 = rdDn(), merge = rdDn();
  Dw me_c1 = rdDw(), me_f1 = rdDw(), me_f2 = rdDw(), me_cv = rdDw();
  Dw g_z = rdDw(), g_r = rdDw(), g_q = rdDw();
  Dw fh_c1 = rdDw(), fh_c2 = rdDw();
  Dn fh_fc = rdDn();

  // ---- workspace bump allocator ----
  char* wp = (char*)d_ws;
  auto af = [&](long nelem) { float* p = (float*)wp;
                              wp += ((nelem * 4 + 255) & ~255L); return p; };
  auto ah = [&](long nelem) { _Float16* p = (_Float16*)wp;
                              wp += ((nelem * 2 + 255) & ~255L); return p; };
  auto ai = [&](long nelem) { int* p = (int*)wp;
                              wp += ((nelem * 4 + 255) & ~255L); return p; };

  const int L[4] = {2048, 1024, 512, 256};

  float* x1_1 = af(2L * 3 * 4096);
  float* x1_2 = af(2L * 3 * 2048);
  float* x2_1 = af(2L * 3 * 4096);
  float* x2l[4]; for (int i = 0; i < 4; ++i) x2l[i] = af(2L * 3 * L[i]);

  float* t_a = af(2L * 128 * 8192);
  float* t_b = af(2L * 128 * 8192);
  int*   idxE = ai(2L * 4096 * 16);
  float* wkE  = af(2L * 4096 * 16 * 16);
  float* aggE = af(2L * 1536 * 4096);

  // packed operand staging (f16)
  _Float16* xpack = ah(2L * 4096 * 1536);   // max (B,N,Cp)
  _Float16* wpack = ah(256L * 2048);        // max (Op,Cp)
  _Float16* f1p   = ah(2L * 2048 * 128);
  _Float16* f2p   = ah(2L * 2048 * 128);

  float* feat1 = af(2L * 128 * 2048);
  float* feat2 = af(2L * 128 * 2048);
  float* featcr = af(2L * 128 * 2048);
  float* featc = af(2L * 256 * 2048);
  float* hbuf = af(2L * 128 * 2048);
  float* xbuf = af(2L * 128 * 2048);

  int*   knn32 = ai(2L * 2048 * 32);
  float* rel32 = af(2L * 2048 * 32 * 3);

  float* pyr[4];
  for (int i = 0; i < 4; ++i) pyr[i] = af(2L * 2048 * L[i]);
  int* idx3 = ai(2L * 2048 * 3);

  float* xyz2w[4]; for (int i = 0; i < 4; ++i) xyz2w[i] = af(2L * 3 * L[i]);
  int*   idxC = ai(2L * 2048 * 16);
  float* cost = af(2L * 128 * 2048);
  float* corr = af(2L * 128 * 2048);
  float* sbuf = af(2L * 384 * 2048);
  float* cf = af(2L * 128 * 2048);
  float* ff1 = af(2L * 32 * 2048);
  float* ff2 = af(2L * 16 * 2048);
  float* cat144 = af(2L * 144 * 2048);
  float* m125 = af(2L * 125 * 2048);
  float* mfeat = af(2L * 128 * 2048);
  float* hx = af(2L * 384 * 2048);
  float* qin = af(2L * 384 * 2048);
  float* zb = af(2L * 128 * 2048);
  float* rb = af(2L * 128 * 2048);
  float* qb = af(2L * 128 * 2048);
  float* fh1 = af(2L * 128 * 2048);
  float* fh2 = af(2L * 64 * 2048);
  float* dflow = af(2L * 3 * 2048);
  float* flow = af(2L * 3 * 2048);

  // ---- launch helpers ----
  auto conv1 = [&](int act, const Dn& p, const float* x, float* out,
                   int C, int N, int O) {
    int Cp = (C + 31) & ~31;
    int Op = (O + 31) & ~31;
    pack_x_kernel<<<nblk((long)NB * N * Cp, 256), 256, 0, stream>>>(x, C, N, Cp, xpack);
    pack_w_kernel<<<nblk((long)Op * Cp, 256), 256, 0, stream>>>(p.w, C, O, Cp, Op, wpack);
    dim3 g(N / 32, Op / 32, NB), blk(32);
    if      (act == 0) conv1_wmma_kernel<0><<<g, blk, 0, stream>>>(wpack, p.b, xpack, out, Cp, N, O);
    else if (act == 1) conv1_wmma_kernel<1><<<g, blk, 0, stream>>>(wpack, p.b, xpack, out, Cp, N, O);
    else if (act == 2) conv1_wmma_kernel<2><<<g, blk, 0, stream>>>(wpack, p.b, xpack, out, Cp, N, O);
    else if (act == 3) conv1_wmma_kernel<3><<<g, blk, 0, stream>>>(wpack, p.b, xpack, out, Cp, N, O);
    else               conv1_wmma_kernel<4><<<g, blk, 0, stream>>>(wpack, p.b, xpack, out, Cp, N, O);
  };
  auto copy_ch = [&](float* dst, int Cd, int doff, const float* src, int Cs,
                     int soff, int Cc, int N) {
    copy_ch_kernel<<<nblk((long)NB * Cc * N, 256), 256, 0, stream>>>(dst, Cd, doff, src, Cs, soff, Cc, N);
  };
  auto dw_s = [&](const Dw& p, const float* feat, int C, int kuse, float* s) {
    dw_s_kernel<<<nblk((long)NB * C * 2048, 256), 256, 0, stream>>>(
        rel32, knn32, feat, p.wn.w, p.wn.b, C, kuse, 2048, s);
  };
  auto run_pconv = [&](const Pc& P, const float* src, int Ns, const float* feat,
                       int C, const float* qry, int Nq, int Cout, float* out) {
    knn_kernel<16><<<nblk((long)NB * Nq, 128), 128, 0, stream>>>(src, Ns, qry, Nq, idxE);
    pconv_wk_kernel<<<nblk((long)NB * Nq * 16, 256), 256, 0, stream>>>(
        src, Ns, qry, Nq, idxE, P.wn.w, P.wn.b, wkE);
    pconv_agg_kernel<<<nblk((long)NB * C * 16 * Nq, 256), 256, 0, stream>>>(
        feat, C, Ns, idxE, Nq, wkE, aggE);
    conv1(2, P.lin, aggE, out, C * 16, Nq, Cout);
  };
  auto run_encoder = [&](const EncW& E, const float* xyz0, const float* xyz1lvl,
                         const float* xyz2lvl, float* out) {
    conv1(1, E.l0[0], xyz0, t_a, 3, 8192, 64);
    conv1(1, E.l0[1], t_a, t_b, 64, 8192, 64);
    conv1(1, E.m0[0], t_b, t_a, 64, 8192, 64);
    conv1(1, E.m0[1], t_a, t_b, 64, 8192, 96);
    run_pconv(E.c0, xyz0, 8192, t_b, 96, xyz1lvl, 4096, 96, t_a);   // t_a: (2,96,4096)
    conv1(1, E.m1[0], t_a, t_b, 96, 4096, 96);
    conv1(1, E.m1[1], t_b, t_a, 96, 4096, 128);
    run_pconv(E.c1, xyz1lvl, 4096, t_a, 128, xyz2lvl, 2048, 128, out);
  };

  // ---- stage 0: compacted coordinate slices ----
  slice_xyz_kernel<<<nblk(2L * 3 * 4096, 256), 256, 0, stream>>>(pc1, 8192, 4096, x1_1);
  slice_xyz_kernel<<<nblk(2L * 3 * 2048, 256), 256, 0, stream>>>(pc1, 8192, 2048, x1_2);
  slice_xyz_kernel<<<nblk(2L * 3 * 4096, 256), 256, 0, stream>>>(pc2, 8192, 4096, x2_1);
  for (int i = 0; i < 4; ++i)
    slice_xyz_kernel<<<nblk(2L * 3 * L[i], 256), 256, 0, stream>>>(pc2, 8192, L[i], x2l[i]);
  zero_kernel<<<nblk(2L * 3 * 2048, 256), 256, 0, stream>>>(flow, 2 * 3 * 2048);

  // ---- encoders ----
  run_encoder(fnet, pc1, x1_1, x1_2, feat1);
  run_encoder(fnet, pc2, x2_1, x2l[0], feat2);
  run_encoder(cnet, pc1, x1_1, x1_2, featcr);
  conv1(0, aligner, featcr, featc, 128, 2048, 256);
  split_featc_kernel<<<nblk(2L * 128 * 2048, 256), 256, 0, stream>>>(featc, hbuf, xbuf, 2048);

  // ---- fixed self-KNN geometry of xyz1 ----
  knn_kernel<32><<<nblk(2L * 2048, 128), 128, 0, stream>>>(x1_2, 2048, x1_2, 2048, knn32);
  relpre_kernel<<<nblk(2L * 2048 * 32, 256), 256, 0, stream>>>(x1_2, knn32, rel32, 2048);

  // ---- correlation pyramid (WMMA on packed feats) ----
  pack_x_kernel<<<nblk(2L * 2048 * 128, 256), 256, 0, stream>>>(feat1, 128, 2048, 128, f1p);
  pack_x_kernel<<<nblk(2L * 2048 * 128, 256), 256, 0, stream>>>(feat2, 128, 2048, 128, f2p);
  corrvol_wmma_kernel<<<dim3(2048 / 32, 2048 / 32, NB), 32, 0, stream>>>(
      f1p, f2p, 2048, 2048, 128, 1.f / 128.f, pyr[0]);
  for (int i = 1; i < 4; ++i) {
    knn_kernel<3><<<nblk((long)NB * L[i], 128), 128, 0, stream>>>(x2l[i - 1], L[i - 1], x2l[i], L[i], idx3);
    pyr_avg_kernel<<<nblk((long)NB * 2048 * L[i], 256), 256, 0, stream>>>(pyr[i - 1], L[i - 1], idx3, L[i], pyr[i]);
  }

  // ---- GRU iterations ----
  for (int it = 0; it < 4; ++it) {
    // backwarp (flow==0 at it 0 -> identity)
    for (int i = 0; i < 4; ++i)
      knn3_interp_kernel<<<nblk((long)NB * L[i], 128), 128, 0, stream>>>(
          x1_2, 2048, x2l[i], L[i], flow, 3, x2l[i], xyz2w[i]);
    // correlation
    for (int i = 0; i < 4; ++i) {
      knn_kernel<16><<<nblk(2L * 2048, 128), 128, 0, stream>>>(xyz2w[i], L[i], x1_2, 2048, idxC);
      cost_kernel<<<nblk(2L * 2048, 128), 128, 0, stream>>>(
          x1_2, xyz2w[i], L[i], idxC, pyr[i], cm0.w, cm0.b, cm1.w, cm1.b, i, cost);
    }
    conv1(2, merge, cost, corr, 128, 2048, 128);
    // motion encoder
    dw_s(me_c1, corr, 128, 16, sbuf); conv1(2, me_c1.out, sbuf, cf, 128, 2048, 128);
    dw_s(me_f1, flow, 3, 32, sbuf);   conv1(2, me_f1.out, sbuf, ff1, 3, 2048, 32);
    dw_s(me_f2, ff1, 32, 16, sbuf);   conv1(2, me_f2.out, sbuf, ff2, 32, 2048, 16);
    copy_ch(cat144, 144, 0, cf, 128, 0, 128, 2048);
    copy_ch(cat144, 144, 128, ff2, 16, 0, 16, 2048);
    dw_s(me_cv, cat144, 144, 16, sbuf); conv1(2, me_cv.out, sbuf, m125, 144, 2048, 125);
    copy_ch(mfeat, 128, 0, m125, 125, 0, 125, 2048);
    copy_ch(mfeat, 128, 125, flow, 3, 0, 3, 2048);
    // hx = [h, x, mfeat]
    copy_ch(hx, 384, 0, hbuf, 128, 0, 128, 2048);
    copy_ch(hx, 384, 128, xbuf, 128, 0, 128, 2048);
    copy_ch(hx, 384, 256, mfeat, 128, 0, 128, 2048);
    dw_s(g_z, hx, 384, 4, sbuf); conv1(4, g_z.out, sbuf, zb, 384, 2048, 128);
    dw_s(g_r, hx, 384, 4, sbuf); conv1(4, g_r.out, sbuf, rb, 384, 2048, 128);
    // qin = [r*h, x, mfeat]
    rh_kernel<<<nblk(2L * 128 * 2048, 256), 256, 0, stream>>>(rb, hbuf, qin, 2048);
    copy_ch(qin, 384, 128, xbuf, 128, 0, 128, 2048);
    copy_ch(qin, 384, 256, mfeat, 128, 0, 128, 2048);
    dw_s(g_q, qin, 384, 4, sbuf); conv1(3, g_q.out, sbuf, qb, 384, 2048, 128);
    gru_combine_kernel<<<nblk(2L * 128 * 2048, 256), 256, 0, stream>>>(hbuf, zb, qb, 2048);
    // flow head
    dw_s(fh_c1, hbuf, 128, 32, sbuf); conv1(2, fh_c1.out, sbuf, fh1, 128, 2048, 128);
    dw_s(fh_c2, fh1, 128, 32, sbuf);  conv1(2, fh_c2.out, sbuf, fh2, 128, 2048, 64);
    conv1(0, fh_fc, fh2, dflow, 64, 2048, 3);
    add_kernel<<<nblk(2L * 3 * 2048, 256), 256, 0, stream>>>(flow, dflow, 2 * 3 * 2048);
    // upsample flow to full cloud -> output slice
    knn3_interp_kernel<<<nblk(2L * 8192, 128), 128, 0, stream>>>(
        x1_2, 2048, pc1, 8192, flow, 3, nullptr,
        (float*)d_out + (long)it * 2 * 3 * 8192);
  }
}
